// CrossAttentionLayer_70729521430969
// MI455X (gfx1250) — matrix operand order repaired and checked by reading, compiled, not run on hardware
//
#include <hip/hip_runtime.h>
#include <cstdint>

// CDNA5 / gfx1250: wave32, WMMA 16x16x32 f16 -> f32 accumulate.
typedef __attribute__((ext_vector_type(16))) _Float16 v16h;
typedef __attribute__((ext_vector_type(8)))  float    v8f;

#define H_DIM  256
#define H_DIV  128
#define HEADS  4
#define NSEG   64
#define LMAXP  512

static __device__ inline v8f wmma_f32_f16(v16h a, v16h b, v8f c) {
  // 8 args: (neg_a, A, neg_b, B, c_mod, C, reuse_a, reuse_b)
  return __builtin_amdgcn_wmma_f32_16x16x32_f16(false, a, false, b, (short)0, c,
                                                false, false);
}

// ---------------------------------------------------------------------------
// Prefix sums of segment counts -> offsets (trivial, 64 segments)
// ---------------------------------------------------------------------------
__global__ void prefix_kernel(const int* __restrict__ scnt,
                              const int* __restrict__ dcnt,
                              int* __restrict__ soff, int* __restrict__ doff) {
  if (threadIdx.x == 0 && blockIdx.x == 0) {
    int s = 0, d = 0;
    for (int i = 0; i < NSEG; ++i) {
      soff[i] = s; s += scnt[i];
      doff[i] = d; d += dcnt[i];
    }
    soff[NSEG] = s; doff[NSEG] = d;
  }
}

// ---------------------------------------------------------------------------
// Generic WMMA GEMM: C[M,N] = epi( [A0|A1][M,K0+K1] @ W[N,K]^T + bias )
// One wave (32 threads) computes a 16(M) x 64(N) tile.
//   mode 0: + bias
//   mode 1: BN(eval) then ReLU
//   mode 2: BN(eval) then + resid (residual stride == N)
// Requires: M%16==0, N%64==0, K%32==0, K0%32==0.
// ---------------------------------------------------------------------------
__global__ __launch_bounds__(32) void gemm_epilogue_kernel(
    const float* __restrict__ A0, int K0,
    const float* __restrict__ A1, int K1,
    const float* __restrict__ W,
    const float* __restrict__ bias,
    const float* __restrict__ gam, const float* __restrict__ bet,
    const float* __restrict__ rmean, const float* __restrict__ rvar,
    const float* __restrict__ resid,
    float* __restrict__ C, int N, int mode)
{
  const int K    = K0 + K1;
  const int lane = threadIdx.x;
  const int hh   = lane >> 4;   // lane half (0/1)
  const int lm   = lane & 15;
  const int r0   = blockIdx.x << 4;
  const int n0   = blockIdx.y << 6;

  v8f acc[4] = {};

  for (int k0 = 0; k0 < K; k0 += 32) {
    // ---- A fragment: 16x32 f16; lane half selects K+8, vgpr pair layout ----
    const float* arow;
    if (k0 < K0) arow = A0 + (size_t)(r0 + lm) * K0 + k0;
    else         arow = A1 + (size_t)(r0 + lm) * K1 + (k0 - K0);
    v16h af;
#pragma unroll
    for (int j = 0; j < 8; ++j) {
      const int kk = ((j >> 2) << 4) + (hh << 3) + ((j & 3) << 1);
      const float2 p = *reinterpret_cast<const float2*>(arow + kk);
      af[2 * j]     = (_Float16)p.x;
      af[2 * j + 1] = (_Float16)p.y;
    }
    // ---- 4 B fragments (32x16 each): B[k][n] = W[n][k] ----
#pragma unroll
    for (int nt = 0; nt < 4; ++nt) {
      const float* wrow = W + (size_t)(n0 + (nt << 4) + lm) * K + k0;
      v16h bf;
#pragma unroll
      for (int j = 0; j < 8; ++j) {
        const int kk = ((j >> 2) << 4) + (hh << 3) + ((j & 3) << 1);
        const float2 p = *reinterpret_cast<const float2*>(wrow + kk);
        bf[2 * j]     = (_Float16)p.x;
        bf[2 * j + 1] = (_Float16)p.y;
      }
      acc[nt] = wmma_f32_f16(af, bf, acc[nt]);
    }
  }

  // ---- epilogue (C/D layout: vgpr r -> M=r+8*half, N=lane%16) ----
#pragma unroll
  for (int nt = 0; nt < 4; ++nt) {
    const int n   = n0 + (nt << 4) + lm;
    const float bia = bias[n];
    float sc = 1.f, sh = 0.f;
    if (mode >= 1) {
      const float inv = rsqrtf(rvar[n] + 1e-5f) * gam[n];
      sc = inv;
      sh = bet[n] - rmean[n] * inv;
    }
#pragma unroll
    for (int r = 0; r < 8; ++r) {
      const int m = r + (hh << 3);
      float val = acc[nt][r] + bia;
      if (mode >= 1) val = val * sc + sh;
      if (mode == 1) val = fmaxf(val, 0.f);
      if (mode == 2) val += resid[(size_t)(r0 + m) * N + n];
      C[(size_t)(r0 + m) * N + n] = val;
    }
  }
}

// ---------------------------------------------------------------------------
// Attention: one wave per (16-query tile, head, segment).
// Channel layout per torch view: channel c = d*HEADS + h, d in [0,32).
// S = Q K^T / sqrt(32) -> LDS, exact softmax over valid Ls keys, O = P V.
// ---------------------------------------------------------------------------
__global__ __launch_bounds__(32) void attn_kernel(
    const float* __restrict__ q,     // [Nd, 128]
    const float* __restrict__ kbuf,  // [Ns, 128]
    const float* __restrict__ vbuf,  // [Ns, 128]
    const int* __restrict__ soff, const int* __restrict__ doff,
    const int* __restrict__ scnt, const int* __restrict__ dcnt,
    float* __restrict__ att)         // [Nd, 128] (may alias q)
{
  __shared__ float S[16][LMAXP + 8];

  const int qt = blockIdx.x;
  const int h  = blockIdx.y;
  const int b  = blockIdx.z;
  const int Ld = dcnt[b];
  if ((qt << 4) >= Ld) return;
  const int Ls   = scnt[b];
  const int so   = soff[b];
  const int dofs = doff[b];
  const int lane = threadIdx.x;
  const int hh   = lane >> 4;
  const int lm   = lane & 15;

  // ---- Q fragment: A-matrix 16x32 over head-dim d (c = d*4 + h) ----
  int qr = (qt << 4) + lm;
  if (qr >= Ld) qr = Ld - 1;                 // clamped rows never stored
  const float* qp = q + (size_t)(dofs + qr) * H_DIV + h;
  v16h qf;
#pragma unroll
  for (int j = 0; j < 8; ++j) {
    const int d0 = ((j >> 2) << 4) + (hh << 3) + ((j & 3) << 1);
    qf[2 * j]     = (_Float16)qp[(size_t)d0 * HEADS];
    qf[2 * j + 1] = (_Float16)qp[(size_t)(d0 + 1) * HEADS];
  }

  // ---- scores S[q, key] into LDS ----
  const float scale = 0.1767766952966369f;   // 1/sqrt(32)
  const int KT = (Ls + 15) >> 4;
  for (int kt = 0; kt < KT; ++kt) {
    int key = (kt << 4) + lm;
    if (key >= Ls) key = Ls - 1;
    const float* kp = kbuf + (size_t)(so + key) * H_DIV + h;
    v16h kf;                                  // B[d][key] = K[key][d]
#pragma unroll
    for (int j = 0; j < 8; ++j) {
      const int d0 = ((j >> 2) << 4) + (hh << 3) + ((j & 3) << 1);
      kf[2 * j]     = (_Float16)kp[(size_t)d0 * HEADS];
      kf[2 * j + 1] = (_Float16)kp[(size_t)(d0 + 1) * HEADS];
    }
    v8f sc = {};
    sc = wmma_f32_f16(qf, kf, sc);
#pragma unroll
    for (int r = 0; r < 8; ++r)
      S[r + (hh << 3)][(kt << 4) + lm] = sc[r] * scale;
  }
  __threadfence_block();

  // ---- exact softmax over valid keys (padding -> prob 0, matches -1e9 mask)
  const int Kpad = ((Ls + 31) >> 5) << 5;
  if (lane < 16) {
    float mx = -1e30f;
    for (int i = 0; i < Ls; ++i) mx = fmaxf(mx, S[lane][i]);
    float sum = 0.f;
    for (int i = 0; i < Ls; ++i) sum += __expf(S[lane][i] - mx);
    const float inv = 1.f / sum;
    for (int i = 0; i < Ls; ++i) S[lane][i] = __expf(S[lane][i] - mx) * inv;
    for (int i = Ls; i < Kpad; ++i) S[lane][i] = 0.f;
  }
  __threadfence_block();

  // ---- O = P @ V, key tiles of 32 ----
  v8f o[2] = {};
  for (int kt = 0; kt < (Kpad >> 5); ++kt) {
    v16h pf;                                   // A fragment of P (16x32)
#pragma unroll
    for (int j = 0; j < 8; ++j) {
      const int kk = ((j >> 2) << 4) + (hh << 3) + ((j & 3) << 1);
      pf[2 * j]     = (_Float16)S[lm][(kt << 5) + kk];
      pf[2 * j + 1] = (_Float16)S[lm][(kt << 5) + kk + 1];
    }
#pragma unroll
    for (int nt = 0; nt < 2; ++nt) {
      const int d = (nt << 4) + lm;
      v16h vf;                                 // B[key][d] = V[key][d]
#pragma unroll
      for (int j = 0; j < 8; ++j) {
        const int kl = ((j >> 2) << 4) + (hh << 3) + ((j & 3) << 1);
        int key0 = (kt << 5) + kl;
        int key1 = key0 + 1;
        if (key0 >= Ls) key0 = Ls - 1;         // P==0 there, value irrelevant
        if (key1 >= Ls) key1 = Ls - 1;
        vf[2 * j]     = (_Float16)vbuf[(size_t)(so + key0) * H_DIV + (size_t)d * HEADS + h];
        vf[2 * j + 1] = (_Float16)vbuf[(size_t)(so + key1) * H_DIV + (size_t)d * HEADS + h];
      }
      o[nt] = wmma_f32_f16(pf, vf, o[nt]);
    }
  }

  // ---- store packed msg (pre-merge) ----
#pragma unroll
  for (int nt = 0; nt < 2; ++nt) {
#pragma unroll
    for (int r = 0; r < 8; ++r) {
      const int m  = r + (hh << 3);
      const int qg = (qt << 4) + m;
      if (qg < Ld) {
        const int d = (nt << 4) + lm;
        att[(size_t)(dofs + qg) * H_DIV + (size_t)d * HEADS + h] = o[nt][r];
      }
    }
  }
}

// ---------------------------------------------------------------------------
extern "C" void kernel_launch(void* const* d_in, const int* in_sizes, int n_in,
                              void* d_out, int out_size, void* d_ws, size_t ws_size,
                              hipStream_t stream) {
  (void)n_in; (void)out_size; (void)ws_size;

  const float* src_h = (const float*)d_in[0];
  const float* dst_h = (const float*)d_in[1];
  const int*   scnt  = (const int*)d_in[2];
  const int*   dcnt  = (const int*)d_in[3];
  const float* Wq  = (const float*)d_in[4];  const float* bq  = (const float*)d_in[5];
  const float* Wk  = (const float*)d_in[6];  const float* bk  = (const float*)d_in[7];
  const float* Wv  = (const float*)d_in[8];  const float* bv  = (const float*)d_in[9];
  const float* Wm  = (const float*)d_in[10]; const float* bm  = (const float*)d_in[11];
  const float* W1  = (const float*)d_in[12]; const float* b1  = (const float*)d_in[13];
  const float* g1  = (const float*)d_in[14]; const float* be1 = (const float*)d_in[15];
  const float* rm1 = (const float*)d_in[16]; const float* rv1 = (const float*)d_in[17];
  const float* W2  = (const float*)d_in[18]; const float* b2  = (const float*)d_in[19];
  const float* g2  = (const float*)d_in[20]; const float* be2 = (const float*)d_in[21];
  const float* rm2 = (const float*)d_in[22]; const float* rv2 = (const float*)d_in[23];

  const int Ms = in_sizes[0] / H_DIM;   // packed src tokens (24576)
  const int Md = in_sizes[1] / H_DIM;   // packed dst tokens (24576)

  // workspace layout: [ints: soff(80) doff(80) pad][floats...]
  int*   soff = (int*)d_ws;
  int*   doff = soff + 80;
  float* fws  = (float*)d_ws + 256;
  float* qb   = fws;                                  // [Md,128] (reused as att)
  float* kb   = qb  + (size_t)Md * H_DIV;             // [Ms,128]
  float* vb   = kb  + (size_t)Ms * H_DIV;             // [Ms,128]
  float* msg  = vb  + (size_t)Ms * H_DIV;             // [Md,128]
  float* x1   = msg + (size_t)Md * H_DIV;             // [Md,256]
  float* attb = qb;                                   // alias (disjoint per-block RMW)

  prefix_kernel<<<dim3(1), dim3(32), 0, stream>>>(scnt, dcnt, soff, doff);

  // Q = dst_h @ Wq^T + bq ; K,V from src_h
  gemm_epilogue_kernel<<<dim3(Md / 16, H_DIV / 64), dim3(32), 0, stream>>>(
      dst_h, H_DIM, nullptr, 0, Wq, bq,
      nullptr, nullptr, nullptr, nullptr, nullptr, qb, H_DIV, 0);
  gemm_epilogue_kernel<<<dim3(Ms / 16, H_DIV / 64), dim3(32), 0, stream>>>(
      src_h, H_DIM, nullptr, 0, Wk, bk,
      nullptr, nullptr, nullptr, nullptr, nullptr, kb, H_DIV, 0);
  gemm_epilogue_kernel<<<dim3(Ms / 16, H_DIV / 64), dim3(32), 0, stream>>>(
      src_h, H_DIM, nullptr, 0, Wv, bv,
      nullptr, nullptr, nullptr, nullptr, nullptr, vb, H_DIV, 0);

  // per-segment multi-head cross attention
  attn_kernel<<<dim3(LMAXP / 16, HEADS, NSEG), dim3(32), 0, stream>>>(
      qb, kb, vb, soff, doff, scnt, dcnt, attb);

  // merge conv: msg = att @ Wm^T + bm
  gemm_epilogue_kernel<<<dim3(Md / 16, H_DIV / 64), dim3(32), 0, stream>>>(
      attb, H_DIV, nullptr, 0, Wm, bm,
      nullptr, nullptr, nullptr, nullptr, nullptr, msg, H_DIV, 0);

  // x1 = relu(bn1([dst_h | msg] @ W1^T + b1))
  gemm_epilogue_kernel<<<dim3(Md / 16, H_DIM / 64), dim3(32), 0, stream>>>(
      dst_h, H_DIM, msg, H_DIV, W1, b1,
      g1, be1, rm1, rv1, nullptr, x1, H_DIM, 1);

  // out = dst_h + bn2(x1 @ W2^T + b2)
  gemm_epilogue_kernel<<<dim3(Md / 16, H_DIM / 64), dim3(32), 0, stream>>>(
      x1, H_DIM, nullptr, 0, W2, b2,
      g2, be2, rm2, rv2, dst_h, (float*)d_out, H_DIM, 2);
}
